// NaNaiveSWinHGNNet_30305289240777
// MI455X (gfx1250) — matrix operand order repaired and checked by reading, not compiled, not run on hardware
//
#include <hip/hip_runtime.h>

// Problem constants from the reference: B=4, N=2048, D=512, Hd=128, k=10.
#define BB  4
#define NN  2048
#define DD  512
#define HD  128
#define KNN 10
#define NEG_SLOPE 0.01f

typedef __attribute__((ext_vector_type(8)))  float  v8f;
typedef __attribute__((ext_vector_type(8)))  __bf16 v8bf;
typedef __attribute__((ext_vector_type(16))) __bf16 v16bf;

// ---------------------------------------------------------------- utilities
__global__ void zero_f32(float* __restrict__ p, int n) {
  int i = blockIdx.x * blockDim.x + threadIdx.x;
  int stride = gridDim.x * blockDim.x;
  for (; i < n; i += stride) p[i] = 0.0f;
}

// Split fp32 -> bf16 hi/lo pair:  x == hi + lo (+ O(2^-18) rel)
__global__ void split_bf16(const float* __restrict__ x, __bf16* __restrict__ xh,
                           __bf16* __restrict__ xl, int n) {
  int i = blockIdx.x * blockDim.x + threadIdx.x;
  int stride = gridDim.x * blockDim.x;
  for (; i < n; i += stride) {
    float v = x[i];
    __bf16 h = (__bf16)v;
    xh[i] = h;
    xl[i] = (__bf16)(v - (float)h);
  }
}

// W[D,HD] -> Wt[HD,D] split into bf16 hi/lo (contiguous-K B fragments).
__global__ void split_transpose_w(const float* __restrict__ W, __bf16* __restrict__ wth,
                                  __bf16* __restrict__ wtl) {
  int t = blockIdx.x * blockDim.x + threadIdx.x;     // t = n*DD + k
  if (t >= HD * DD) return;
  int n = t / DD, k = t - n * DD;
  float v = W[(size_t)k * HD + n];
  __bf16 h = (__bf16)v;
  wth[t] = h;
  wtl[t] = (__bf16)(v - (float)h);
}

// One wave32 per row: sq[row] = sum_k x[row,k]^2 (fp32, matches reference).
__global__ void row_norms(const float* __restrict__ x, float* __restrict__ sq) {
  const int lane = threadIdx.x & 31;
  const int wave = threadIdx.x >> 5;
  const int row  = blockIdx.x * 8 + wave;            // row in [0, BB*NN)
  const float* xr = x + (size_t)row * DD;
  float s = 0.0f;
  for (int k = lane; k < DD; k += 32) { float v = xr[k]; s += v * v; }
  #pragma unroll
  for (int off = 16; off; off >>= 1) s += __shfl_xor(s, off, 32);
  if (lane == 0) sq[row] = s;
}

// ------------------------------------------------ bf16 WMMA fragment loads
// A 16x32 bf16 layout (per lane M=l15, hi=lane>>4):
//   VGPR0-7: K = k0 + 16*hi + 0..15 (contiguous, matches B)
__device__ __forceinline__ v16bf load_a_frag(const __bf16* rowp, int k0, int hi) {
  return *(const v16bf*)(rowp + k0 + 16 * hi);
}
// B 32x16 bf16 layout (per lane N=l15): VGPR0-7 hold K = k0 + 16*hi + 0..15
__device__ __forceinline__ v16bf load_b_frag(const __bf16* colp, int k0, int hi) {
  return *(const v16bf*)(colp + k0 + 16 * hi);
}

#define WMMA_BF16(A, B, C) \
  __builtin_amdgcn_wmma_f32_16x16x32_bf16(false, (A), false, (B), (short)0, (C), false, false)

// 3-term split-bf16 accumulation: C += Ah*Bh + Ah*Bl + Al*Bh  (~fp32 accurate)
#define SPLIT_MAC(Ah, Al, Bh, Bl, C)  \
  do { C = WMMA_BF16(Ah, Bh, C); C = WMMA_BF16(Ah, Bl, C); C = WMMA_BF16(Al, Bh, C); } while (0)

// gfx1250 async copy global -> LDS (16B per lane), tracked by ASYNCcnt.
__device__ __forceinline__ void async_copy_b128(unsigned lds_addr, const void* gptr) {
  asm volatile("global_load_async_to_lds_b128 %0, %1, off"
               :: "v"(lds_addr), "v"(gptr) : "memory");
}
__device__ __forceinline__ void async_wait0() {
  asm volatile("s_wait_asynccnt 0" ::: "memory");
}

// ------------------------------- d2 = sq_i + sq_j - 2 * x @ x^T  (bf16 WMMA)
// Each wave: 32x32 output (2x2 tiles of 16x16). 8 waves/block -> 256x32/block.
// The 32-column B K-panel (hi+lo, 64KB) is async-staged into LDS once and
// shared by all 8 waves.  grid = (NN/32, NN/256), block = 256, 1 batch/launch.
__device__ __forceinline__ void store_d2_tile(float* __restrict__ d2,
                                              const float* __restrict__ sqb,
                                              int rowT, int colT, int l15, int hi, v8f acc) {
  const float sqj = sqb[colT + l15];
  #pragma unroll
  for (int r = 0; r < 8; ++r) {
    const int row = rowT + r + 8 * hi;
    d2[(size_t)row * NN + colT + l15] = sqb[row] + sqj - 2.0f * acc[r];
  }
}

__global__ void dist_wmma_bf16(const __bf16* __restrict__ xh, const __bf16* __restrict__ xl,
                               const float* __restrict__ sqb, float* __restrict__ d2) {
  __shared__ __bf16 ldsB[2 * 32 * DD];               // [panel hi|lo][colRel][k] = 64KB
  const int t    = threadIdx.x;
  const int lane = t & 31;
  const int wave = t >> 5;
  const int hi = lane >> 4, l15 = lane & 15;
  const int colBase = blockIdx.x * 32;
  const int rowBase = (blockIdx.y * 8 + wave) * 32;

  // ---- async-stage the B panel: 4096 x 16B chunks, 16 per thread ---------
  {
    const unsigned ldsBase = (unsigned)(size_t)(void*)ldsB; // addr[31:0] = LDS offset
    #pragma unroll
    for (int i = 0; i < 16; ++i) {
      const int c     = t + 256 * i;                 // chunk id 0..4095
      const int panel = c >> 11;                     // 0: hi, 1: lo
      const int cc    = c & 2047;
      const int col   = cc >> 6;                     // 64 x 16B chunks per column
      const int kb    = (cc & 63) << 4;              // byte offset within row
      const __bf16* g = (panel ? xl : xh) + (size_t)(colBase + col) * DD;
      async_copy_b128(ldsBase + (unsigned)(panel * 32768 + col * (DD * 2) + kb),
                      (const char*)g + kb);
    }
    async_wait0();          // drain this wave's ASYNCcnt
    __syncthreads();        // make all waves' staged data visible
  }

  const __bf16* a0h = xh + (size_t)(rowBase + l15) * DD;
  const __bf16* a1h = a0h + (size_t)16 * DD;
  const __bf16* a0l = xl + (size_t)(rowBase + l15) * DD;
  const __bf16* a1l = a0l + (size_t)16 * DD;
  // B fragments now come from LDS (per lane: column = l15 / l15+16)
  const __bf16* b0h = &ldsB[(size_t)l15 * DD];
  const __bf16* b1h = b0h + (size_t)16 * DD;
  const __bf16* b0l = &ldsB[32 * DD + (size_t)l15 * DD];
  const __bf16* b1l = b0l + (size_t)16 * DD;

  v8f acc00 = {}, acc01 = {}, acc10 = {}, acc11 = {};
  for (int k0 = 0; k0 < DD; k0 += 32) {
    v16bf A0h = load_a_frag(a0h, k0, hi), A0l = load_a_frag(a0l, k0, hi);
    v16bf A1h = load_a_frag(a1h, k0, hi), A1l = load_a_frag(a1l, k0, hi);
    v16bf B0h = load_b_frag(b0h, k0, hi), B0l = load_b_frag(b0l, k0, hi);
    v16bf B1h = load_b_frag(b1h, k0, hi), B1l = load_b_frag(b1l, k0, hi);
    SPLIT_MAC(A0h, A0l, B0h, B0l, acc00);
    SPLIT_MAC(A0h, A0l, B1h, B1l, acc01);
    SPLIT_MAC(A1h, A1l, B0h, B0l, acc10);
    SPLIT_MAC(A1h, A1l, B1h, B1l, acc11);
  }
  store_d2_tile(d2, sqb, rowBase,      colBase,      l15, hi, acc00);
  store_d2_tile(d2, sqb, rowBase,      colBase + 16, l15, hi, acc01);
  store_d2_tile(d2, sqb, rowBase + 16, colBase,      l15, hi, acc10);
  store_d2_tile(d2, sqb, rowBase + 16, colBase + 16, l15, hi, acc11);
}

// ------------------------------------------ y = x @ W + b   (bf16 WMMA, 2x2)
// grid = (HD/32, NN/256, BB), block = 256.
__global__ void linear_wmma_bf16(const __bf16* __restrict__ xh, const __bf16* __restrict__ xl,
                                 const __bf16* __restrict__ wth, const __bf16* __restrict__ wtl,
                                 const float* __restrict__ bias, float* __restrict__ y) {
  const int lane = threadIdx.x & 31;
  const int wave = threadIdx.x >> 5;
  const int hi = lane >> 4, l15 = lane & 15;
  const int b = blockIdx.z;
  const int colBase = blockIdx.x * 32;
  const int rowBase = (blockIdx.y * 8 + wave) * 32;

  const __bf16* a0h = xh + ((size_t)b * NN + rowBase + l15) * DD;
  const __bf16* a1h = a0h + (size_t)16 * DD;
  const __bf16* a0l = xl + ((size_t)b * NN + rowBase + l15) * DD;
  const __bf16* a1l = a0l + (size_t)16 * DD;
  const __bf16* b0h = wth + (size_t)(colBase + l15) * DD;   // Wt[col][k]
  const __bf16* b1h = b0h + (size_t)16 * DD;
  const __bf16* b0l = wtl + (size_t)(colBase + l15) * DD;
  const __bf16* b1l = b0l + (size_t)16 * DD;

  const float bv0 = bias[colBase + l15];
  const float bv1 = bias[colBase + 16 + l15];
  v8f acc00 = { bv0,bv0,bv0,bv0,bv0,bv0,bv0,bv0 };
  v8f acc01 = { bv1,bv1,bv1,bv1,bv1,bv1,bv1,bv1 };
  v8f acc10 = acc00, acc11 = acc01;

  for (int k0 = 0; k0 < DD; k0 += 32) {
    v16bf A0h = load_a_frag(a0h, k0, hi), A0l = load_a_frag(a0l, k0, hi);
    v16bf A1h = load_a_frag(a1h, k0, hi), A1l = load_a_frag(a1l, k0, hi);
    v16bf B0h = load_b_frag(b0h, k0, hi), B0l = load_b_frag(b0l, k0, hi);
    v16bf B1h = load_b_frag(b1h, k0, hi), B1l = load_b_frag(b1l, k0, hi);
    SPLIT_MAC(A0h, A0l, B0h, B0l, acc00);
    SPLIT_MAC(A0h, A0l, B1h, B1l, acc01);
    SPLIT_MAC(A1h, A1l, B0h, B0l, acc10);
    SPLIT_MAC(A1h, A1l, B1h, B1l, acc11);
  }

  float* yb = y + (size_t)b * NN * HD;
  #pragma unroll
  for (int r = 0; r < 8; ++r) {
    const int r0 = rowBase + r + 8 * hi;
    const int r1 = r0 + 16;
    yb[(size_t)r0 * HD + colBase + l15]      = acc00[r];
    yb[(size_t)r0 * HD + colBase + 16 + l15] = acc01[r];
    yb[(size_t)r1 * HD + colBase + l15]      = acc10[r];
    yb[(size_t)r1 * HD + colBase + 16 + l15] = acc11[r];
  }
}

// --------------------------- per-row top-k (k=10 smallest d2) + Dv scatter
__global__ void topk_knn(const float* __restrict__ d2, int* __restrict__ idxb,
                         float* __restrict__ Dvb) {
  const int e = blockIdx.x * blockDim.x + threadIdx.x;
  if (e >= NN) return;
  const float* row = d2 + (size_t)e * NN;
  float bd[KNN]; int bi[KNN];
  #pragma unroll
  for (int j = 0; j < KNN; ++j) { bd[j] = 3.4e38f; bi[j] = 0; }
  for (int c = 0; c < NN; ++c) {
    const float v = row[c];
    if (v < bd[KNN - 1]) {
      int p = KNN - 1;
      while (p > 0 && v < bd[p - 1]) { bd[p] = bd[p - 1]; bi[p] = bi[p - 1]; --p; }
      bd[p] = v; bi[p] = c;
    }
  }
  #pragma unroll
  for (int j = 0; j < KNN; ++j) {
    idxb[e * KNN + j] = bi[j];
    atomicAdd(&Dvb[bi[j]], 1.0f);
  }
}

// ---------- m[e] = (1/De) * sum_{v in knn(e)} y[v] * Dv[v]^-1/2   (De == 10)
__global__ void gather_edges(const float* __restrict__ y, const int* __restrict__ idx,
                             const float* __restrict__ Dv, float* __restrict__ m) {
  const int e = blockIdx.x, b = blockIdx.y, d = threadIdx.x;
  const int*   ie  = idx + ((size_t)b * NN + e) * KNN;
  const float* yb  = y   + (size_t)b * NN * HD;
  const float* Dvb = Dv  + (size_t)b * NN;
  float s = 0.0f;
  #pragma unroll
  for (int j = 0; j < KNN; ++j) {
    const int v = ie[j];
    s += yb[(size_t)v * HD + d] * rsqrtf(fmaxf(Dvb[v], 1.0f));
  }
  m[((size_t)b * NN + e) * HD + d] = s * (1.0f / KNN);
}

// ---------------- acc[v] += m[e] for every edge e with v in knn(e) (scatter)
__global__ void scatter_edges(const float* __restrict__ m, const int* __restrict__ idx,
                              float* __restrict__ acc) {
  const int e = blockIdx.x, b = blockIdx.y, d = threadIdx.x;
  const int* ie = idx + ((size_t)b * NN + e) * KNN;
  const float mv = m[((size_t)b * NN + e) * HD + d];
  float* ab = acc + (size_t)b * NN * HD;
  #pragma unroll
  for (int j = 0; j < KNN; ++j)
    atomicAdd(&ab[(size_t)ie[j] * HD + d], mv);
}

// --------- out[b, level*HD + d] = mean_v leaky( Dv[v]^-1/2 * acc[b,v,d] )
__global__ void finalize_mean(const float* __restrict__ acc, const float* __restrict__ Dv,
                              float* __restrict__ out, int level) {
  const int b = blockIdx.x, d = threadIdx.x;
  const float* ab  = acc + (size_t)b * NN * HD;
  const float* Dvb = Dv  + (size_t)b * NN;
  float s = 0.0f;
  for (int v = 0; v < NN; ++v) {
    const float h = ab[(size_t)v * HD + d] * rsqrtf(fmaxf(Dvb[v], 1.0f));
    s += (h >= 0.0f) ? h : NEG_SLOPE * h;
  }
  out[(size_t)b * (2 * HD) + level * HD + d] = s * (1.0f / NN);
}

// ---------------------------------------------------------------------------
extern "C" void kernel_launch(void* const* d_in, const int* in_sizes, int n_in,
                              void* d_out, int out_size, void* d_ws, size_t ws_size,
                              hipStream_t stream) {
  (void)in_sizes; (void)n_in; (void)out_size; (void)ws_size;
  // inputs: x0, x1, c0, c1, W0, b0, W1, b1
  char* w = (char*)d_ws;
  size_t off = 0;
  auto carve = [&](size_t bytes) -> char* {
    char* p = w + off;
    off += (bytes + 255) & ~(size_t)255;
    return p;
  };
  float*  sq  = (float*) carve(sizeof(float)  * BB * NN);
  float*  Dv  = (float*) carve(sizeof(float)  * BB * NN);
  float*  y   = (float*) carve(sizeof(float)  * (size_t)BB * NN * HD);
  float*  m   = (float*) carve(sizeof(float)  * (size_t)BB * NN * HD);
  float*  acc = (float*) carve(sizeof(float)  * (size_t)BB * NN * HD);
  float*  d2  = (float*) carve(sizeof(float)  * (size_t)NN * NN);     // reused per batch
  int*    idx = (int*)   carve(sizeof(int)    * (size_t)BB * NN * KNN);
  __bf16* xh  = (__bf16*)carve(sizeof(__bf16) * (size_t)BB * NN * DD);
  __bf16* xl  = (__bf16*)carve(sizeof(__bf16) * (size_t)BB * NN * DD);
  __bf16* wth = (__bf16*)carve(sizeof(__bf16) * (size_t)HD * DD);
  __bf16* wtl = (__bf16*)carve(sizeof(__bf16) * (size_t)HD * DD);

  float* out = (float*)d_out;

  for (int lvl = 0; lvl < 2; ++lvl) {
    const float* x    = (const float*)d_in[lvl];          // x0 or x1
    const float* W    = (const float*)d_in[4 + 2 * lvl];  // W0 or W1
    const float* bias = (const float*)d_in[5 + 2 * lvl];  // b0 or b1

    zero_f32<<<64,   256, 0, stream>>>(Dv,  BB * NN);
    zero_f32<<<1024, 256, 0, stream>>>(acc, BB * NN * HD);

    split_bf16<<<2048, 256, 0, stream>>>(x, xh, xl, BB * NN * DD);
    split_transpose_w<<<(HD * DD) / 256, 256, 0, stream>>>(W, wth, wtl);

    row_norms<<<BB * NN / 8, 256, 0, stream>>>(x, sq);
    linear_wmma_bf16<<<dim3(HD / 32, NN / 256, BB), 256, 0, stream>>>(xh, xl, wth, wtl,
                                                                      bias, y);

    for (int b = 0; b < BB; ++b) {
      dist_wmma_bf16<<<dim3(NN / 32, NN / 256), 256, 0, stream>>>(
          xh + (size_t)b * NN * DD, xl + (size_t)b * NN * DD, sq + (size_t)b * NN, d2);
      topk_knn<<<NN / 256, 256, 0, stream>>>(d2, idx + (size_t)b * NN * KNN,
                                             Dv + (size_t)b * NN);
    }

    gather_edges <<<dim3(NN, BB), HD, 0, stream>>>(y, idx, Dv, m);
    scatter_edges<<<dim3(NN, BB), HD, 0, stream>>>(m, idx, acc);
    finalize_mean<<<BB, HD, 0, stream>>>(acc, Dv, out, lvl);
  }
}
